// HybridQNN_53970559041820
// MI455X (gfx1250) — compile-verified, matrix operand order checked by reading, not run on hardware
//
#include <hip/hip_runtime.h>
#include <hip/hip_bf16.h>
#include <math.h>

typedef __attribute__((ext_vector_type(16))) _Float16 v16h;
typedef __attribute__((ext_vector_type(8)))  _Float16 v8h;
typedef __attribute__((ext_vector_type(8)))  float    v8f;
typedef __attribute__((ext_vector_type(4)))  int      v4i;
typedef __attribute__((address_space(1))) v4i as1_v4i;   // global int4
typedef __attribute__((address_space(3))) v4i as3_v4i;   // LDS int4

#define WPB 8  // waves per block (256 threads, wave32)

// ---------------------------------------------------------------------------
// NCHW f32 -> NHWC f16   (out[((b*H+h)*W+w)*C + c])
__global__ void k_nchw_to_nhwc_f16(const float* __restrict__ in, _Float16* __restrict__ out,
                                   int C, int HW, int n) {
  int i = blockIdx.x * blockDim.x + threadIdx.x;
  if (i >= n) return;
  int c = i % C;
  int pix = i / C;           // b*HW + hw
  int b = pix / HW, hw = pix % HW;
  out[i] = (_Float16)in[((long)b * C + c) * HW + hw];
}

// conv weight OIHW -> f16 [K][Cout], K = (kh*3+kw)*Cin + cin
__global__ void k_wT_conv(const float* __restrict__ in, _Float16* __restrict__ out,
                          int Cin, int Cout) {
  int i = blockIdx.x * blockDim.x + threadIdx.x;
  int Ktot = Cin * 9;
  if (i >= Ktot * Cout) return;
  int k = i / Cout, n = i % Cout;
  int blk = k / Cin, cin = k % Cin;          // blk = kh*3+kw
  out[i] = (_Float16)in[(long)n * Ktot + cin * 9 + blk];
}

// fc1 weight [128][16384(c,h,w)] -> f16 [K][128] with K in NHWC order (h,w,c)
__global__ void k_wT_fc1(const float* __restrict__ in, _Float16* __restrict__ out) {
  int i = blockIdx.x * blockDim.x + threadIdx.x;
  if (i >= 16384 * 128) return;
  int k = i / 128, n = i % 128;
  int hw = k / 64, c = k % 64;               // NHWC k -> (hw, c)
  out[i] = (_Float16)in[(long)n * 16384 + c * 256 + hw];
}

// plain transpose + pad: out[k*Npad+n] = (n<N) ? in[n*K+k] : 0
__global__ void k_wT_f16(const float* __restrict__ in, _Float16* __restrict__ out,
                         int N, int K, int Npad) {
  int i = blockIdx.x * blockDim.x + threadIdx.x;
  if (i >= K * Npad) return;
  int k = i / Npad, n = i % Npad;
  out[i] = (n < N) ? (_Float16)in[n * K + k] : (_Float16)0.f;
}

__global__ void k_zero(float* __restrict__ p, int n) {
  int i = blockIdx.x * blockDim.x + threadIdx.x;
  if (i < n) p[i] = 0.f;
}

// ---------------------------------------------------------------------------
// implicit-GEMM 3x3 SAME conv, NHWC, WMMA f16 -> f32, fused per-channel sum/sumsq.
// out = conv(in)+bias (NHWC f32); st[c] += sum, st[Cout+c] += sumsq (batch-norm stats)
template<int Cin, int Cout, int H, int W>
__global__ __launch_bounds__(256) void k_conv3x3_wmma(
    const _Float16* __restrict__ in,   // [B][H][W][Cin] f16
    const _Float16* __restrict__ wk,   // [(kh*3+kw)*Cin+cin][Cout] f16
    const float*    __restrict__ bias, // [Cout]
    float*          __restrict__ out,  // [B][H][W][Cout] f32
    float*          __restrict__ st)   // [2*Cout] sum / sumsq (pre-zeroed)
{
  constexpr int HW   = H * W;
  constexpr int Ktot = Cin * 9;
  __shared__ __align__(16) _Float16 bs[32 * 16];
  __shared__ float ssum[16], ssq[16];

  const int lane = threadIdx.x & 31;
  const int wv   = threadIdx.x >> 5;
  const int l15  = lane & 15;
  const int mbase = (blockIdx.x * WPB + wv) * 16;
  const int nbase = blockIdx.y * 16;
  const int pix = mbase + l15;
  const int b   = pix / HW;
  const int hwp = pix % HW;
  const int ph  = hwp / W;
  const int pw  = hwp % W;
  const int klo = (lane < 16) ? 0 : 8;

  if (threadIdx.x < 16) { ssum[threadIdx.x] = 0.f; ssq[threadIdx.x] = 0.f; }

  v8f acc;
  {
    float bv = bias[nbase + l15];
#pragma unroll
    for (int r = 0; r < 8; r++) acc[r] = bv;
  }

  for (int kb = 0; kb < Ktot; kb += 32) {
    __syncthreads();
    // stage B chunk [32][16] in LDS (same nbase for all 8 waves)
#if __has_builtin(__builtin_amdgcn_global_load_async_to_lds_b128)
    if constexpr (Ktot % 32 == 0) {
      // async DMA: 64 lanes x 16B, tracked by ASYNCcnt
      if (threadIdx.x < 64) {
        int k = threadIdx.x >> 1, half = threadIdx.x & 1;
        const _Float16* src = wk + (long)(kb + k) * Cout + nbase + half * 8;
        __builtin_amdgcn_global_load_async_to_lds_b128(
            (as1_v4i*)(void*)src,
            (as3_v4i*)(void*)&bs[k * 16 + half * 8],
            0, 0);
#if __has_builtin(__builtin_amdgcn_s_wait_asynccnt)
        __builtin_amdgcn_s_wait_asynccnt(0);
#else
        asm volatile("s_wait_asynccnt 0x0" ::: "memory");
#endif
      }
    } else {
      for (int t = threadIdx.x; t < 512; t += 256) {
        int kk = t >> 4, nn = t & 15;
        int kg = kb + kk;
        bs[t] = (kg < Ktot) ? wk[(long)kg * Cout + nbase + nn] : (_Float16)0.f;
      }
    }
#else
    for (int t = threadIdx.x; t < 512; t += 256) {
      int kk = t >> 4, nn = t & 15;
      int kg = kb + kk;
      bs[t] = (kg < Ktot) ? wk[(long)kg * Cout + nbase + nn] : (_Float16)0.f;
    }
#endif
    if (kb + 32 < Ktot)
      __builtin_prefetch(wk + (long)(kb + 32) * Cout + nbase, 0, 2);
    __syncthreads();

    v16h a, bf;
    if constexpr (Cin % 32 == 0) {
      // whole chunk lies in one (kh,kw) tap: one bounds check, two 16B loads
      const int blk  = kb / Cin;                 // kh*3+kw (constant over chunk)
      const int cin0 = (kb % Cin) + klo;
      const int ih = ph + blk / 3 - 1, iw = pw + blk % 3 - 1;
      v8h lo = {}, hi = {};
      if (ih >= 0 && ih < H && iw >= 0 && iw < W) {
        const _Float16* p = in + ((long)((b * H + ih) * W + iw)) * Cin + cin0;
        lo = *(const v8h*)p;         // cin0 .. cin0+7
        hi = *(const v8h*)(p + 16);  // cin0+16 .. cin0+23
      }
      a = __builtin_shufflevector(lo, hi, 0,1,2,3,4,5,6,7,8,9,10,11,12,13,14,15);
    } else {
      // generic gather (conv1, Cin=3)
#pragma unroll
      for (int e = 0; e < 16; e++) {
        const int kk = klo + e + ((e >= 8) ? 8 : 0);
        const int K  = kb + kk;
        _Float16 v = (_Float16)0.f;
        if (K < Ktot) {
          int blk = K / Cin, cin = K % Cin;
          int ih = ph + blk / 3 - 1, iw = pw + blk % 3 - 1;
          if (ih >= 0 && ih < H && iw >= 0 && iw < W)
            v = in[((long)((b * H + ih) * W + iw)) * Cin + cin];
        }
        a[e] = v;
      }
    }
#pragma unroll
    for (int e = 0; e < 16; e++) {
      const int kk = klo + e + ((e >= 8) ? 8 : 0);
      bf[e] = bs[kk * 16 + l15];
    }
    acc = __builtin_amdgcn_wmma_f32_16x16x32_f16(false, a, false, bf,
                                                 (short)0, acc, false, false);
  }

  const int nch = nbase + l15;
  float s1 = 0.f, s2 = 0.f;
#pragma unroll
  for (int r = 0; r < 8; r++) {
    int mm = r + ((lane < 16) ? 0 : 8);
    float v = acc[r];
    out[(long)(mbase + mm) * Cout + nch] = v;   // NHWC: coalesced in N
    s1 += v; s2 += v * v;
  }
  // fused BN statistics: register -> LDS atomics -> 32 global atomics per block
  atomicAdd(&ssum[l15], s1);
  atomicAdd(&ssq[l15], s2);
  __syncthreads();
  if (threadIdx.x < 16) {
    atomicAdd(&st[nbase + threadIdx.x], ssum[threadIdx.x]);
    atomicAdd(&st[Cout + nbase + threadIdx.x], ssq[threadIdx.x]);
  }
}

// ---------------------------------------------------------------------------
// fused BN(batch stats) + ReLU + f16, NHWC
__global__ void k_bn_relu_f16(const float* __restrict__ x, const float* __restrict__ st,
                              const float* __restrict__ g, const float* __restrict__ bt,
                              _Float16* __restrict__ out, int C, int n, float invN)
{
  int i = blockIdx.x * blockDim.x + threadIdx.x;
  if (i >= n) return;
  int c = i % C;
  float m   = st[c] * invN;
  float var = st[C + c] * invN - m * m;
  float sc  = g[c] * rsqrtf(var + 1e-5f);
  float sh  = bt[c] - m * sc;
  out[i] = (_Float16)fmaxf(x[i] * sc + sh, 0.f);
}

// fused BN + ReLU + 2x2 maxpool + f16, NHWC
__global__ void k_bn_relu_pool_f16(const float* __restrict__ x, const float* __restrict__ st,
                                   const float* __restrict__ g, const float* __restrict__ bt,
                                   _Float16* __restrict__ out, int C, int H, int W, int n, float invN)
{
  int i = blockIdx.x * blockDim.x + threadIdx.x;
  if (i >= n) return;
  int Wo = W >> 1, Ho = H >> 1;
  int c = i % C;  int t = i / C;
  int wo = t % Wo; t /= Wo;
  int ho = t % Ho; int bb = t / Ho;
  float m   = st[c] * invN;
  float var = st[C + c] * invN - m * m;
  float sc  = g[c] * rsqrtf(var + 1e-5f);
  float sh  = bt[c] - m * sc;
  const float* p = x + ((long)((bb * H + 2 * ho) * W + 2 * wo)) * C + c;
  float v0 = fmaxf(p[0] * sc + sh, 0.f);
  float v1 = fmaxf(p[C] * sc + sh, 0.f);
  float v2 = fmaxf(p[(long)W * C] * sc + sh, 0.f);
  float v3 = fmaxf(p[(long)(W + 1) * C] * sc + sh, 0.f);
  out[i] = (_Float16)fmaxf(fmaxf(v0, v1), fmaxf(v2, v3));
}

// ---------------------------------------------------------------------------
// FC GEMM: out[M][Nreal] = A[M][K](f16) @ Bw[K][Npad](f16) + bias, optional ReLU
__global__ __launch_bounds__(256) void k_fc_wmma(
    const _Float16* __restrict__ A, const _Float16* __restrict__ Bw,
    const float* __restrict__ bias, float* __restrict__ out32, _Float16* __restrict__ out16,
    int M, int K, int Npad, int Nreal, int ldo, int relu)
{
  int lane = threadIdx.x & 31, wv = threadIdx.x >> 5;
  int ntiles = Npad >> 4;
  int tile = blockIdx.x * WPB + wv;
  int mt = tile / ntiles, nt = tile % ntiles;
  if (mt * 16 >= M) return;
  int l15 = lane & 15;
  int klo = (lane < 16) ? 0 : 8;
  long row = mt * 16 + l15;

  v8f acc = {};
  for (int kb = 0; kb < K; kb += 32) {
    v8h lo = *(const v8h*)(A + row * K + kb + klo);
    v8h hi = *(const v8h*)(A + row * K + kb + klo + 16);
    v16h a = __builtin_shufflevector(lo, hi, 0,1,2,3,4,5,6,7,8,9,10,11,12,13,14,15);
    v16h bf;
#pragma unroll
    for (int e = 0; e < 16; e++) {
      int kk = klo + e + ((e >= 8) ? 8 : 0);
      bf[e] = Bw[(long)(kb + kk) * Npad + nt * 16 + l15];
    }
    acc = __builtin_amdgcn_wmma_f32_16x16x32_f16(false, a, false, bf,
                                                 (short)0, acc, false, false);
  }

  int col = nt * 16 + l15;
  float bv = (col < Nreal) ? bias[col] : 0.f;
#pragma unroll
  for (int r = 0; r < 8; r++) {
    int mm = r + ((lane < 16) ? 0 : 8);
    long ro = mt * 16 + mm;
    float v = acc[r] + bv;
    if (relu) v = fmaxf(v, 0.f);
    if (col < Nreal) {
      out32[ro * ldo + col] = v;
      if (out16) out16[ro * ldo + col] = (_Float16)v;
    }
  }
}

// ---------------------------------------------------------------------------
// 10-qubit state-vector sim (state in LDS) + expval(Z_w) + head GEMV
__device__ __forceinline__ float2 cmul(float2 a, float2 b) {
  return make_float2(a.x * b.x - a.y * b.y, a.x * b.y + a.y * b.x);
}
__device__ __forceinline__ float2 cadd(float2 a, float2 b) {
  return make_float2(a.x + b.x, a.y + b.y);
}

__global__ __launch_bounds__(256) void k_quantum_head(
    const float* __restrict__ xq,  // [B][10]
    const float* __restrict__ qw,  // [6][10][3]
    const float* __restrict__ hwt, // [10][10]
    const float* __restrict__ hb,  // [10]
    float* __restrict__ out)       // [B][10]
{
  __shared__ float2 s0[1024];
  __shared__ float2 s1[1024];
  __shared__ float  red[256];
  __shared__ float  evs[16];
  const int b = blockIdx.x;
  const int tid = threadIdx.x;

  for (int i = tid; i < 1024; i += 256) s0[i] = make_float2(i == 0 ? 1.f : 0.f, 0.f);
  __syncthreads();

  // embedding: RY(pi * xq[b,w]) on wire w  (wire w <-> bit 9-w)
  for (int w = 0; w < 10; w++) {
    float th = 0.5f * 3.14159265358979323846f * xq[b * 10 + w];
    float cw = cosf(th), sw = sinf(th);
    int sh = 9 - w, right = 1 << sh;
    for (int p = tid; p < 512; p += 256) {
      int i0 = ((p >> sh) << (sh + 1)) | (p & (right - 1));
      int i1 = i0 | right;
      float2 a0 = s0[i0], a1 = s0[i1];
      s0[i0] = make_float2(cw * a0.x - sw * a1.x, cw * a0.y - sw * a1.y);
      s0[i1] = make_float2(sw * a0.x + cw * a1.x, sw * a0.y + cw * a1.y);
    }
    __syncthreads();
  }

  // 6 layers: 10 Rot(phi,theta,omega) gates, then CNOT-ring permutation
  for (int l = 0; l < 6; l++) {
    for (int w = 0; w < 10; w++) {
      const float* q = qw + (l * 10 + w) * 3;
      float phi = q[0], th = q[1], om = q[2];
      float c = cosf(0.5f * th), s = sinf(0.5f * th);
      float apo = 0.5f * (phi + om), amo = 0.5f * (phi - om);
      float2 m00 = make_float2( cosf(apo) * c, -sinf(apo) * c);
      float2 m01 = make_float2(-cosf(amo) * s, -sinf(amo) * s);
      float2 m10 = make_float2( cosf(amo) * s, -sinf(amo) * s);
      float2 m11 = make_float2( cosf(apo) * c,  sinf(apo) * c);
      int sh = 9 - w, right = 1 << sh;
      for (int p = tid; p < 512; p += 256) {
        int i0 = ((p >> sh) << (sh + 1)) | (p & (right - 1));
        int i1 = i0 | right;
        float2 a0 = s0[i0], a1 = s0[i1];
        s0[i0] = cadd(cmul(m00, a0), cmul(m01, a1));
        s0[i1] = cadd(cmul(m10, a0), cmul(m11, a1));
      }
      __syncthreads();
    }
    // perm(i) = pg_0(pg_1(...pg_9(i))), pg_w: j ^= ((j>>pc)&1)<<pt
    int r = l % 9 + 1;
    for (int i = tid; i < 1024; i += 256) {
      int j = i;
      for (int w = 9; w >= 0; w--) {
        int pc = 9 - w;
        int pt = 9 - ((w + r) % 10);
        j ^= ((j >> pc) & 1) << pt;
      }
      s1[i] = s0[j];
    }
    __syncthreads();
    for (int i = tid; i < 1024; i += 256) s0[i] = s1[i];
    __syncthreads();
  }

  // expvals <Z_w> = sum_i |amp_i|^2 * (1 - 2*bit_{9-w}(i))
  float myp[4];
#pragma unroll
  for (int t = 0; t < 4; t++) {
    int i = tid + t * 256;
    float2 a = s0[i];
    myp[t] = a.x * a.x + a.y * a.y;
  }
  for (int w = 0; w < 10; w++) {
    float e = 0.f;
#pragma unroll
    for (int t = 0; t < 4; t++) {
      int i = tid + t * 256;
      e += ((i >> (9 - w)) & 1) ? -myp[t] : myp[t];
    }
    red[tid] = e; __syncthreads();
    for (int o = 128; o > 0; o >>= 1) {
      if (tid < o) red[tid] += red[tid + o];
      __syncthreads();
    }
    if (tid == 0) evs[w] = red[0];
    __syncthreads();
  }

  if (tid < 10) {
    float acc = hb[tid];
    for (int w = 0; w < 10; w++) acc += evs[w] * hwt[tid * 10 + w];
    out[b * 10 + tid] = acc;
  }
}

// ---------------------------------------------------------------------------
extern "C" void kernel_launch(void* const* d_in, const int* in_sizes, int n_in,
                              void* d_out, int out_size, void* d_ws, size_t ws_size,
                              hipStream_t stream) {
  const float* x    = (const float*)d_in[0];
  const float* c1w  = (const float*)d_in[1];  const float* c1b = (const float*)d_in[2];
  const float* g1   = (const float*)d_in[3];  const float* b1  = (const float*)d_in[4];
  const float* c2w  = (const float*)d_in[5];  const float* c2b = (const float*)d_in[6];
  const float* g2   = (const float*)d_in[7];  const float* b2  = (const float*)d_in[8];
  const float* c3w  = (const float*)d_in[9];  const float* c3b = (const float*)d_in[10];
  const float* g3   = (const float*)d_in[11]; const float* b3  = (const float*)d_in[12];
  const float* c4w  = (const float*)d_in[13]; const float* c4b = (const float*)d_in[14];
  const float* g4   = (const float*)d_in[15]; const float* b4  = (const float*)d_in[16];
  const float* f1w  = (const float*)d_in[17]; const float* f1b = (const float*)d_in[18];
  const float* f2w  = (const float*)d_in[19]; const float* f2b = (const float*)d_in[20];
  const float* qw   = (const float*)d_in[21];
  const float* hwt  = (const float*)d_in[22]; const float* hb  = (const float*)d_in[23];
  float* outp = (float*)d_out;

  // workspace carve-up
  char* base = (char*)d_ws;
  size_t off = 0;
  auto carve = [&](size_t bytes) { char* p = base + off; off = (off + bytes + 255) & ~(size_t)255; return p; };
  float*     raw    = (float*)    carve((size_t)33554432 * 4); // raw conv out f32 NHWC (max)
  _Float16*  actA   = (_Float16*) carve((size_t)33554432 * 2); // f16 NHWC activations ping
  _Float16*  actB   = (_Float16*) carve((size_t)33554432 * 2); // f16 NHWC activations pong
  _Float16*  wconv  = (_Float16*) carve((size_t)36864 * 2);    // conv weights f16 [K][Cout]
  _Float16*  wfc1   = (_Float16*) carve((size_t)16384 * 128 * 2);
  _Float16*  wfc2   = (_Float16*) carve((size_t)128 * 16 * 2);
  float*     stats  = (float*)    carve(128 * 4);              // per-channel sum / sumsq
  _Float16*  fc1h   = (_Float16*) carve((size_t)256 * 128 * 2);
  float*     fc1f   = (float*)    carve((size_t)256 * 128 * 4);
  float*     xqb    = (float*)    carve((size_t)256 * 10 * 4);

  const int B = 256;
  auto g1d = [](long n) { return dim3((unsigned)((n + 255) / 256)); };

  // ---- conv1 (3->32, 64x64) ----
  k_nchw_to_nhwc_f16<<<g1d((long)B*3*4096), 256, 0, stream>>>(x, actA, 3, 4096, B*3*4096);
  k_wT_conv<<<g1d(27*32), 256, 0, stream>>>(c1w, wconv, 3, 32);
  k_zero<<<1, 128, 0, stream>>>(stats, 128);
  k_conv3x3_wmma<3,32,64,64><<<dim3(8192, 2), 256, 0, stream>>>(actA, wconv, c1b, raw, stats);
  k_bn_relu_f16<<<g1d((long)B*4096*32), 256, 0, stream>>>(raw, stats, g1, b1, actB, 32,
                                                          B*4096*32, 1.f/(B*4096));
  // ---- conv2 (32->32, 64x64) + pool ----
  k_wT_conv<<<g1d(288*32), 256, 0, stream>>>(c2w, wconv, 32, 32);
  k_zero<<<1, 128, 0, stream>>>(stats, 128);
  k_conv3x3_wmma<32,32,64,64><<<dim3(8192, 2), 256, 0, stream>>>(actB, wconv, c2b, raw, stats);
  k_bn_relu_pool_f16<<<g1d((long)B*1024*32), 256, 0, stream>>>(raw, stats, g2, b2, actA,
                                                               32, 64, 64, B*1024*32, 1.f/(B*4096));
  // ---- conv3 (32->64, 32x32) ----
  k_wT_conv<<<g1d(288*64), 256, 0, stream>>>(c3w, wconv, 32, 64);
  k_zero<<<1, 128, 0, stream>>>(stats, 128);
  k_conv3x3_wmma<32,64,32,32><<<dim3(2048, 4), 256, 0, stream>>>(actA, wconv, c3b, raw, stats);
  k_bn_relu_f16<<<g1d((long)B*1024*64), 256, 0, stream>>>(raw, stats, g3, b3, actB, 64,
                                                          B*1024*64, 1.f/(B*1024));
  // ---- conv4 (64->64, 32x32) + pool ----
  k_wT_conv<<<g1d(576*64), 256, 0, stream>>>(c4w, wconv, 64, 64);
  k_zero<<<1, 128, 0, stream>>>(stats, 128);
  k_conv3x3_wmma<64,64,32,32><<<dim3(2048, 4), 256, 0, stream>>>(actB, wconv, c4b, raw, stats);
  k_bn_relu_pool_f16<<<g1d((long)B*256*64), 256, 0, stream>>>(raw, stats, g4, b4, actA,
                                                              64, 32, 32, B*256*64, 1.f/(B*1024));
  // ---- fc1 (16384 -> 128, K remapped to NHWC order) + relu ----
  k_wT_fc1<<<g1d((long)16384*128), 256, 0, stream>>>(f1w, wfc1);
  k_fc_wmma<<<dim3(16), 256, 0, stream>>>(actA, wfc1, f1b, fc1f, fc1h,
                                          256, 16384, 128, 128, 128, 1);
  // ---- fc2 (128 -> 10) ----
  k_wT_f16<<<g1d(128*16), 256, 0, stream>>>(f2w, wfc2, 10, 128, 16);
  k_fc_wmma<<<dim3(2), 256, 0, stream>>>(fc1h, wfc2, f2b, xqb, (_Float16*)nullptr,
                                         256, 128, 16, 10, 10, 0);
  // ---- quantum circuit + head ----
  k_quantum_head<<<dim3(256), 256, 0, stream>>>(xqb, qw, hwt, hb, outp);
}
